// MultiHeadAttention_16655883173923
// MI455X (gfx1250) — matrix operand order repaired and checked by reading, not compiled, not run on hardware
//
#include <hip/hip_runtime.h>
#include <hip/hip_bf16.h>

// ---- shapes ----
#define BB 8
#define SS 1024
#define DD 512
#define HH 8
#define DQ 64
#define DV 64
#define PP 32

typedef __bf16 v8bf  __attribute__((ext_vector_type(8)));
typedef __bf16 v16bf __attribute__((ext_vector_type(16)));
typedef float  v8f   __attribute__((ext_vector_type(8)));

__device__ inline unsigned short f2bf_bits(float f) {
    unsigned u = __builtin_bit_cast(unsigned, f);
    unsigned r = u + 0x7FFFu + ((u >> 16) & 1u);   // round-to-nearest-even
    return (unsigned short)(r >> 16);
}
__device__ inline __bf16 f2bf(float f) {
    return __builtin_bit_cast(__bf16, f2bf_bits(f));
}

// A-fragment (16x32 bf16, row-major source): p = row_start + kbase + (lane>=16?8:0)
__device__ inline v16bf ld_a16(const unsigned short* p) {
    v8bf lo = *reinterpret_cast<const v8bf*>(p);
    v8bf hi = *reinterpret_cast<const v8bf*>(p + 16);
    return __builtin_shufflevector(lo, hi, 0,1,2,3,4,5,6,7,8,9,10,11,12,13,14,15);
}
// B-fragment (32x16 bf16): p = col_start + kbase + (lane>=16?8:0)
__device__ inline v16bf ld_b16(const unsigned short* p) {
    v8bf lo = *reinterpret_cast<const v8bf*>(p);
    v8bf hi = *reinterpret_cast<const v8bf*>(p + 16);
    return __builtin_shufflevector(lo, hi, 0,1,2,3,4,5,6,7,8,9,10,11,12,13,14,15);
}
// A-fragment from fp32 LDS probs row (convert on the fly)
__device__ inline v16bf ld_a16_f32(const float* p) {
    v16bf r;
#pragma unroll
    for (int j = 0; j < 8; j++) { r[j] = f2bf(p[j]); r[j + 8] = f2bf(p[16 + j]); }
    return r;
}
__device__ inline v8f wmma_bf16(v16bf a, v16bf b, v8f c) {
    return __builtin_amdgcn_wmma_f32_16x16x32_bf16(false, a, false, b, (short)0, c, false, false);
}

// ---------------- 1) LayerNorm -> bf16 ----------------
__global__ __launch_bounds__(256)
void ln_kernel(const float* __restrict__ x, const float* __restrict__ g,
               const float* __restrict__ bvec, unsigned short* __restrict__ out) {
    __shared__ float red[256];
    int row = blockIdx.x, t = threadIdx.x;
    const float* xr = x + (size_t)row * DD;
    float v0 = xr[t], v1 = xr[t + 256];
    red[t] = v0 + v1; __syncthreads();
    for (int o = 128; o > 0; o >>= 1) { if (t < o) red[t] += red[t + o]; __syncthreads(); }
    float mu = red[0] * (1.f / DD); __syncthreads();
    float d0 = v0 - mu, d1 = v1 - mu;
    red[t] = d0 * d0 + d1 * d1; __syncthreads();
    for (int o = 128; o > 0; o >>= 1) { if (t < o) red[t] += red[t + o]; __syncthreads(); }
    float rstd = rsqrtf(red[0] * (1.f / DD) + 1e-5f);
    unsigned short* outr = out + (size_t)row * DD;
    outr[t]       = f2bf_bits(d0 * rstd * g[t] + bvec[t]);
    outr[t + 256] = f2bf_bits(d1 * rstd * g[t + 256] + bvec[t + 256]);
}

// ---------------- 2) transpose + convert 512x512 weight -> bf16 col-major ----------------
__global__ __launch_bounds__(256)
void cvt_t512(const float* __restrict__ W, unsigned short* __restrict__ WT) {
    int idx = blockIdx.x * 256 + threadIdx.x;      // 512*512
    int n = idx >> 9, k = idx & 511;
    WT[(size_t)n * 512 + k] = f2bf_bits(W[(size_t)k * 512 + n]);
}

// ---------------- 3) pos bias: bias[q,k] = pos[q,k,:]@wp + bp, prescaled 1/sqrt(DQ) -------
__global__ __launch_bounds__(256)
void bias_kernel(const float* __restrict__ pos, const float* __restrict__ wp,
                 const float* __restrict__ bp, float* __restrict__ biasS) {
    int idx = blockIdx.x * 256 + threadIdx.x;      // S*S
    const float4* pp  = (const float4*)(pos + (size_t)idx * PP);
    const float4* wv4 = (const float4*)wp;
    float acc = 0.f;
#pragma unroll
    for (int i = 0; i < 8; i++) {
        float4 p = pp[i], w = wv4[i];
        acc += p.x * w.x + p.y * w.y + p.z * w.z + p.w * w.w;
    }
    biasS[idx] = (acc + bp[0]) * 0.125f;           // 1/sqrt(64)
}

// ---------------- 4/6) WMMA GEMM [8192x512] @ [512x512], 16x64 per wave ----------------
// Software-pipelined: fragments for step k+1 are in flight while step k's 4 WMMAs issue.
// mode 0: -> Kmat bf16 [B,H,S,DQ]; mode 1: -> Vt bf16 [B,H,DV,S]; mode 2: -> seq fp32 (+x)
__global__ __launch_bounds__(128)
void gemm_bf16(const unsigned short* __restrict__ A, const unsigned short* __restrict__ Bw,
               const float* __restrict__ bias, const float* __restrict__ xres,
               unsigned short* __restrict__ obf, float* __restrict__ ofp, int mode) {
    int wave = threadIdx.x >> 5, lane = threadIdx.x & 31;
    int tile = blockIdx.x * 4 + wave;              // 512 Mtiles * 8 Nstrips = 4096 waves
    int tm = tile >> 3, ts = tile & 7;
    int rlo = lane & 15, hiw = (lane >= 16);
    int gn0 = ts * 64 + rlo;
    v8f acc[4];
#pragma unroll
    for (int j = 0; j < 4; j++) {
        float bias_n = bias[gn0 + j * 16];
#pragma unroll
        for (int r = 0; r < 8; r++) acc[j][r] = bias_n;
    }
    if (mode == 2) {
#pragma unroll
        for (int j = 0; j < 4; j++)
#pragma unroll
            for (int r = 0; r < 8; r++) {
                int grow = tm * 16 + r + (hiw ? 8 : 0);
                acc[j][r] += xres[(size_t)grow * DD + gn0 + j * 16];
            }
    }
    const unsigned short* arow  = A + (size_t)(tm * 16 + rlo) * DD + (hiw ? 8 : 0);
    const unsigned short* bcol0 = Bw + (size_t)gn0 * DD + (hiw ? 8 : 0);
    // prologue: fragments for kk = 0
    v16bf af = ld_a16(arow);
    v16bf b0 = ld_b16(bcol0);
    v16bf b1 = ld_b16(bcol0 + 16 * DD);
    v16bf b2 = ld_b16(bcol0 + 32 * DD);
    v16bf b3 = ld_b16(bcol0 + 48 * DD);
#pragma unroll
    for (int kk = 0; kk < DD; kk += 32) {
        v16bf ca = af, c0 = b0, c1 = b1, c2 = b2, c3 = b3;
        int kn = kk + 32;
        if (kn < DD) {                              // static under full unroll
            af = ld_a16(arow + kn);
            b0 = ld_b16(bcol0 + kn);
            b1 = ld_b16(bcol0 + 16 * DD + kn);
            b2 = ld_b16(bcol0 + 32 * DD + kn);
            b3 = ld_b16(bcol0 + 48 * DD + kn);
        }
        acc[0] = wmma_bf16(ca, c0, acc[0]);
        acc[1] = wmma_bf16(ca, c1, acc[1]);
        acc[2] = wmma_bf16(ca, c2, acc[2]);
        acc[3] = wmma_bf16(ca, c3, acc[3]);
    }
#pragma unroll
    for (int j = 0; j < 4; j++) {
        int gn = gn0 + j * 16;
#pragma unroll
        for (int r = 0; r < 8; r++) {
            int grow = tm * 16 + r + (hiw ? 8 : 0);
            int bb = grow >> 10, ss = grow & 1023, hh = gn >> 6, dd = gn & 63;
            if (mode == 0)
                obf[((size_t)(bb * HH + hh) * SS + ss) * DQ + dd] = f2bf_bits(acc[j][r]);
            else if (mode == 1)
                obf[((size_t)(bb * HH + hh) * DV + dd) * SS + ss] = f2bf_bits(acc[j][r]);
            else
                ofp[(size_t)grow * DD + gn] = acc[j][r];
        }
    }
}

// ---------------- 5) attention: scores (WMMA) + softmax + probs@V (WMMA) ----------------
// 2 waves per block share one 64KB score buffer (even/odd key tiles, split rows/ntiles).
__global__ __launch_bounds__(64)
void attn_kernel(const unsigned short* __restrict__ Kmat, const unsigned short* __restrict__ Vt,
                 const float* __restrict__ biasS, const float* __restrict__ wp,
                 const int* __restrict__ lens, float* __restrict__ probs,
                 unsigned short* __restrict__ aout) {
    __shared__ __align__(16) float sS[16 * SS];    // 64 KB: scores, then normalized probs
    int qt = blockIdx.x, h = blockIdx.y, b = blockIdx.z;
    int wave = threadIdx.x >> 5, lane = threadIdx.x & 31;
    int rlo = lane & 15, hiw = (lane >= 16);
    float swp = 0.f;
    for (int p = 0; p < PP; p++) swp += wp[p];     // uniform -> scalar loads
    float ascale = swp * 0.04419417382415922f;     // sum(wp) / (sqrt(H)*sqrt(DQ))
    int len = lens[b];
    const unsigned short* Kb = Kmat + (size_t)(b * HH + h) * SS * DQ;
    const unsigned short* arow = Kb + (size_t)(qt * 16 + rlo) * DQ + (hiw ? 8 : 0);
    // query-row A fragments are loop-invariant (DQ = 64 -> exactly two fragments)
    v16bf af0 = ld_a16(arow);
    v16bf af1 = ld_a16(arow + 32);
    // ---- phase 1: scores; wave w handles key tiles kt = w, w+2, ... (pipelined) ----
    const unsigned short* kb0 = Kb + (size_t)rlo * DQ + (hiw ? 8 : 0);
    v16bf p0 = ld_b16(kb0 + (size_t)wave * 16 * DQ);
    v16bf p1 = ld_b16(kb0 + (size_t)wave * 16 * DQ + 32);
    for (int kt = wave; kt < SS / 16; kt += 2) {
        v16bf q0 = p0, q1 = p1;
        if (kt + 2 < SS / 16) {
            const unsigned short* bn = kb0 + (size_t)(kt + 2) * 16 * DQ;
            p0 = ld_b16(bn);
            p1 = ld_b16(bn + 32);
        }
        v8f acc = {};
        acc = wmma_bf16(af0, q0, acc);
        acc = wmma_bf16(af1, q1, acc);
        int col = kt * 16 + rlo;
#pragma unroll
        for (int r = 0; r < 8; r++) {
            int rm = r + (hiw ? 8 : 0);
            float v = acc[r] * ascale + biasS[(size_t)(qt * 16 + rm) * SS + col];
            if (col >= len) v = -1e30f;
            sS[rm * SS + col] = v;
        }
    }
    __syncthreads();
    // ---- softmax: wave w handles rows w*8 .. w*8+7 (coalesced probs store) ----
    float* prow_base = probs + ((size_t)(b * HH + h) * SS + qt * 16) * SS;
    for (int r = wave * 8; r < wave * 8 + 8; r++) {
        float* rp = sS + r * SS;
        float m = -3.0e38f;
        for (int c = lane; c < SS; c += 32) m = fmaxf(m, rp[c]);
        for (int o = 16; o; o >>= 1) m = fmaxf(m, __shfl_xor(m, o));
        float s = 0.f;
        for (int c = lane; c < SS; c += 32) { float e = __expf(rp[c] - m); rp[c] = e; s += e; }
        for (int o = 16; o; o >>= 1) s += __shfl_xor(s, o);
        float inv = 1.f / s;
        float* gp = prow_base + (size_t)r * SS;
        for (int c = lane; c < SS; c += 32) { float pv = rp[c] * inv; rp[c] = pv; gp[c] = pv; }
    }
    __syncthreads();
    // ---- phase 2: out = probs @ V; wave w handles DV columns w*32 .. w*32+31 (pipelined) --
    v8f o0 = {}, o1 = {};
    const float* pArow = sS + rlo * SS;
    const unsigned short* vb = Vt + (size_t)(b * HH + h) * DV * SS + (hiw ? 8 : 0);
    int ab = hiw ? 8 : 0;
    int n0 = wave * 32;
    const unsigned short* v0p = vb + (size_t)(n0 + rlo) * SS;
    const unsigned short* v1p = vb + (size_t)(n0 + 16 + rlo) * SS;
    v16bf af = ld_a16_f32(pArow + ab);
    v16bf w0 = ld_b16(v0p);
    v16bf w1 = ld_b16(v1p);
    for (int mb = 0; mb < SS; mb += 32) {
        v16bf ca = af, c0 = w0, c1 = w1;
        int mn = mb + 32;
        if (mn < SS) {
            af = ld_a16_f32(pArow + mn + ab);
            w0 = ld_b16(v0p + mn);
            w1 = ld_b16(v1p + mn);
        }
        o0 = wmma_bf16(ca, c0, o0);
        o1 = wmma_bf16(ca, c1, o1);
    }
    unsigned short* ao = aout + ((size_t)(b * SS + qt * 16)) * (HH * DV) + h * DV;
#pragma unroll
    for (int r = 0; r < 8; r++) {
        unsigned short* aor = ao + (size_t)(r + (hiw ? 8 : 0)) * (HH * DV);
        aor[n0 + rlo]      = f2bf_bits(o0[r]);
        aor[n0 + 16 + rlo] = f2bf_bits(o1[r]);
    }
}

// ---------------- launcher ----------------
extern "C" void kernel_launch(void* const* d_in, const int* in_sizes, int n_in,
                              void* d_out, int out_size, void* d_ws, size_t ws_size,
                              hipStream_t stream) {
    const float* x    = (const float*)d_in[0];
    const float* ln_g = (const float*)d_in[1];
    const float* ln_b = (const float*)d_in[2];
    const float* Wk   = (const float*)d_in[3];
    const float* bk   = (const float*)d_in[4];
    const float* Wv   = (const float*)d_in[5];
    const float* bv   = (const float*)d_in[6];
    const float* pos  = (const float*)d_in[7];
    const float* wp   = (const float*)d_in[8];
    const float* bp   = (const float*)d_in[9];
    const float* Wo   = (const float*)d_in[10];
    const float* bo   = (const float*)d_in[11];
    const int*   len  = (const int*)d_in[12];

    float* seq   = (float*)d_out;                              // [B,S,D]
    float* probs = seq + (size_t)BB * SS * DD;                 // [B,H,S,S]

    char* w = (char*)d_ws;
    unsigned short* xnb  = (unsigned short*)(w);                       // 8 MB
    unsigned short* wkT  = (unsigned short*)(w + 8388608);             // 512 KB
    unsigned short* wvT  = (unsigned short*)(w + 8912896);
    unsigned short* woT  = (unsigned short*)(w + 9437184);
    unsigned short* kmat = (unsigned short*)(w + 9961472);             // 8 MB
    unsigned short* vt   = (unsigned short*)(w + 18350080);            // 8 MB
    float*          bS   = (float*)(w + 26738688);                     // 4 MB
    unsigned short* aout = (unsigned short*)(w + 30932992);            // 8 MB (total ~39.3 MB)

    ln_kernel  <<<BB * SS, 256, 0, stream>>>(x, ln_g, ln_b, xnb);
    cvt_t512   <<<1024, 256, 0, stream>>>(Wk, wkT);
    cvt_t512   <<<1024, 256, 0, stream>>>(Wv, wvT);
    cvt_t512   <<<1024, 256, 0, stream>>>(Wo, woT);
    bias_kernel<<<SS * SS / 256, 256, 0, stream>>>(pos, wp, bp, bS);
    gemm_bf16  <<<1024, 128, 0, stream>>>(xnb, wkT, bk, nullptr, kmat, nullptr, 0);
    gemm_bf16  <<<1024, 128, 0, stream>>>(xnb, wvT, bv, nullptr, vt, nullptr, 1);
    attn_kernel<<<dim3(SS / 16, HH, BB), 64, 0, stream>>>(kmat, vt, bS, wp, len, probs, aout);
    gemm_bf16  <<<1024, 128, 0, stream>>>(aout, woT, bo, x, nullptr, seq, 2);
}
